// StepAttention_2542620639611
// MI455X (gfx1250) — matrix operand order repaired and checked by reading, not compiled, not run on hardware
//
#include <hip/hip_runtime.h>
#include <math.h>

// ---------------------------------------------------------------------------
// Shapes (fixed by the reference)
// ---------------------------------------------------------------------------
#define BATCH 4
#define SEQ   1024
#define HDIM  4096
#define NH    32
#define NKV   8
#define DHEAD 128
#define QKV_N 6144   // NH*DHEAD + 2*NKV*DHEAD
#define MROWS (BATCH * SEQ)  // 4096

// ---------------------------------------------------------------------------
// WMMA fragment types (gfx1250, wave32): bf16 16x16x32, fp32 accumulate
// ---------------------------------------------------------------------------
typedef __bf16        bf16x16 __attribute__((ext_vector_type(16)));
typedef unsigned int  uintx8  __attribute__((ext_vector_type(8)));
typedef float         floatx8 __attribute__((ext_vector_type(8)));

union Frag {
    bf16x16 b;
    uintx8  u;   // 8 dwords = 16 packed bf16
};

static __device__ inline floatx8 wmma_bf16(const Frag& a, const Frag& b, floatx8 c) {
    return __builtin_amdgcn_wmma_f32_16x16x32_bf16(
        /*neg_a=*/false, a.b, /*neg_b=*/false, b.b,
        /*c_mod=*/(short)0, c, /*reuse_a=*/false, /*reuse_b=*/false);
}

static __device__ inline unsigned short f2bf(float f) {
    unsigned int u = __float_as_uint(f);
    unsigned int r = u + 0x7FFFu + ((u >> 16) & 1u);  // round-to-nearest-even
    return (unsigned short)(r >> 16);
}

// Async global->LDS copy of 16 bytes (GLOBAL_LOAD_ASYNC_TO_LDS_B128, ASYNCcnt).
// LDS dest offset = low 32 bits of the generic LDS pointer (addr[31:0] per ISA).
static __device__ inline void async_copy_b128(void* lds_ptr, const void* gptr) {
    unsigned lds_off = (unsigned)(size_t)lds_ptr;
    unsigned long long ga = (unsigned long long)(size_t)gptr;
    asm volatile("global_load_async_to_lds_b128 %0, %1, off"
                 :: "v"(lds_off), "v"(ga) : "memory");
}
static __device__ inline void wait_async0() {
    asm volatile("s_wait_asynccnt 0x0" ::: "memory");
}

// A-fragment K offsets per dword v (16-bit A 16x32 layout, ISA 7.12.2):
//   v<4 -> K = 2v ;  v>=4 -> K = 16 + 2(v-4) ; lanes 16..31 add +8
#define KA(v) ((v) < 4 ? 2 * (v) : 16 + 2 * ((v)-4))
// B-fragment K offsets per dword v (same K order as A):
//   v<4 -> K = 2v ;  v>=4 -> K = 16 + 2(v-4) ; lanes 16..31 add +8
#define KB(v) ((v) < 4 ? 2 * (v) : 16 + 2 * ((v)-4))

// ---------------------------------------------------------------------------
// Precision prep kernels: run once, so the hot GEMM loops stream bf16 only.
// ---------------------------------------------------------------------------
// fp32 -> bf16 elementwise (hidden states), 4 elems/thread
__global__ __launch_bounds__(256)
void cvt_bf16_kernel(const float* __restrict__ s, unsigned short* __restrict__ d,
                     int n4) {
    int i = blockIdx.x * 256 + threadIdx.x;
    if (i < n4) {
        float4 f = ((const float4*)s)[i];
        unsigned int lo = (unsigned int)f2bf(f.x) | ((unsigned int)f2bf(f.y) << 16);
        unsigned int hi = (unsigned int)f2bf(f.z) | ((unsigned int)f2bf(f.w) << 16);
        ((uint2*)d)[i] = make_uint2(lo, hi);
    }
}

// fp32 [R][C] -> bf16 transposed [C][R] via 32x32 LDS tiles (both sides coalesced)
__global__ __launch_bounds__(256)
void cvt_transpose_kernel(const float* __restrict__ s,
                          unsigned short* __restrict__ d, int R, int C) {
    __shared__ unsigned short t[32][33];
    const int r0 = blockIdx.y * 32, c0 = blockIdx.x * 32;
    const int tx = threadIdx.x & 31, ty = threadIdx.x >> 5;  // 32 x 8
#pragma unroll
    for (int i = 0; i < 32; i += 8)
        t[ty + i][tx] = f2bf(s[(size_t)(r0 + ty + i) * C + c0 + tx]);
    __syncthreads();
#pragma unroll
    for (int i = 0; i < 32; i += 8)
        d[(size_t)(c0 + ty + i) * R + r0 + tx] = t[tx][ty + i];
}

// ---------------------------------------------------------------------------
// bf16 GEMM core: C[128x128 tile] = A[M][K] @ B^T[N][K]   (both bf16, row-major,
// K contiguous).  256 threads = 8 waves; wave tile 16x128; k-step 32.
// Double-buffered LDS staged with async b128 copies; one barrier per k-step.
// LDS row stride 40 elems = 80 B (16B-aligned rows, 20-bank stride -> no
// duplicate banks across the 16 fragment lanes).
// ---------------------------------------------------------------------------
#define GSTRIDE 40

struct GemmAcc { floatx8 acc[8]; };

static __device__ inline void gemm_core(const unsigned short* __restrict__ Abf,
                                        const unsigned short* __restrict__ Bbf,
                                        int Kdim, int mbase, int nbase,
                                        unsigned short (*Al)[128][GSTRIDE],
                                        unsigned short (*Bl)[128][GSTRIDE],
                                        GemmAcc& g) {
    const int tid = threadIdx.x, lane = tid & 31, wv = tid >> 5;
    const int col = lane & 15, khalf = lane >> 4;

#pragma unroll
    for (int t = 0; t < 8; ++t)
#pragma unroll
        for (int v = 0; v < 8; ++v) g.acc[t][v] = 0.0f;

    auto stage = [&](int p, int k0) {
#pragma unroll
        for (int it = 0; it < 2; ++it) {
            int c = it * 256 + tid;                 // 512 chunks of 16B per tile
            int r = c >> 2, c8 = (c & 3) << 3;
            async_copy_b128(&Al[p][r][c8],
                            &Abf[(size_t)(mbase + r) * Kdim + k0 + c8]);
            async_copy_b128(&Bl[p][r][c8],
                            &Bbf[(size_t)(nbase + r) * Kdim + k0 + c8]);
        }
    };

    stage(0, 0);
    int p = 0;
    for (int k0 = 0; k0 < Kdim; k0 += 32) {
        wait_async0();
        __syncthreads();
        if (k0 + 32 < Kdim) stage(p ^ 1, k0 + 32);

        Frag af;
#pragma unroll
        for (int v = 0; v < 8; ++v)
            af.u[v] = *(const unsigned int*)&Al[p][wv * 16 + col][KA(v) + 8 * khalf];
#pragma unroll
        for (int t = 0; t < 8; ++t) {
            Frag bfr;
#pragma unroll
            for (int v = 0; v < 8; ++v)
                bfr.u[v] =
                    *(const unsigned int*)&Bl[p][t * 16 + col][KB(v) + 8 * khalf];
            g.acc[t] = wmma_bf16(af, bfr, g.acc[t]);
        }
        p ^= 1;
    }
}

// ---------------------------------------------------------------------------
// Kernel 1: QKV GEMM -> scatter bf16 into q_ws[b][h][s][d], k/v_ws[b][kvh][s][d]
// ---------------------------------------------------------------------------
__global__ __launch_bounds__(256)
void gemm_qkv_kernel(const unsigned short* __restrict__ Abf,
                     const unsigned short* __restrict__ BT,
                     unsigned short* __restrict__ q_ws,
                     unsigned short* __restrict__ k_ws,
                     unsigned short* __restrict__ v_ws) {
    __shared__ unsigned short Al[2][128][GSTRIDE];
    __shared__ unsigned short Bl[2][128][GSTRIDE];
    const int tid = threadIdx.x, lane = tid & 31, wv = tid >> 5;
    const int col = lane & 15, khalf = lane >> 4;
    const int mbase = blockIdx.y * 128, nbase = blockIdx.x * 128;

    GemmAcc g;
    gemm_core(Abf, BT, HDIM, mbase, nbase, Al, Bl, g);

#pragma unroll
    for (int t = 0; t < 8; ++t) {
        int n = nbase + t * 16 + col;
#pragma unroll
        for (int v = 0; v < 8; ++v) {
            int row = mbase + wv * 16 + v + 8 * khalf;
            int bb = row >> 10, ss = row & 1023;
            unsigned short val = f2bf(g.acc[t][v]);
            int d = n & 127;
            if (n < NH * DHEAD) {
                int hh = n >> 7;
                q_ws[((size_t)((bb * NH + hh) * SEQ + ss)) * DHEAD + d] = val;
            } else if (n < NH * DHEAD + NKV * DHEAD) {
                int hh = (n - NH * DHEAD) >> 7;
                k_ws[((size_t)((bb * NKV + hh) * SEQ + ss)) * DHEAD + d] = val;
            } else {
                int hh = (n - NH * DHEAD - NKV * DHEAD) >> 7;
                v_ws[((size_t)((bb * NKV + hh) * SEQ + ss)) * DHEAD + d] = val;
            }
        }
    }
}

// ---------------------------------------------------------------------------
// Kernel 3: output GEMM -> fp32 d_out
// ---------------------------------------------------------------------------
__global__ __launch_bounds__(256)
void gemm_o_kernel(const unsigned short* __restrict__ Abf,
                   const unsigned short* __restrict__ BT,
                   float* __restrict__ Cout) {
    __shared__ unsigned short Al[2][128][GSTRIDE];
    __shared__ unsigned short Bl[2][128][GSTRIDE];
    const int tid = threadIdx.x, lane = tid & 31, wv = tid >> 5;
    const int col = lane & 15, khalf = lane >> 4;
    const int mbase = blockIdx.y * 128, nbase = blockIdx.x * 128;

    GemmAcc g;
    gemm_core(Abf, BT, HDIM, mbase, nbase, Al, Bl, g);

#pragma unroll
    for (int t = 0; t < 8; ++t) {
        int n = nbase + t * 16 + col;
#pragma unroll
        for (int v = 0; v < 8; ++v) {
            int row = mbase + wv * 16 + v + 8 * khalf;
            Cout[(size_t)row * HDIM + n] = g.acc[t][v];
        }
    }
}

// ---------------------------------------------------------------------------
// Kernel 2: flash attention with ALiBi-sqrt bias + causal mask (GQA 4:1).
// Block = 128 threads = 4 waves; one (b,h) per block; wave w -> query tile
// qt = qt4*4 + w.  K staged via async b128 (row-major, stride 136 elems so
// 16B rows + 68-bank stride); V transposed in-register; P re-layout via LDS.
// ---------------------------------------------------------------------------
__global__ __launch_bounds__(128)
void attn_kernel(const unsigned short* __restrict__ q_ws,
                 const unsigned short* __restrict__ k_ws,
                 const unsigned short* __restrict__ v_ws,
                 unsigned short* __restrict__ o_ws) {
    __shared__ unsigned short Kl[32][136];    // [key][dim]
    __shared__ unsigned short Vl[128][34];    // [dim][key]
    __shared__ unsigned short Pl[4][16][34];  // per-wave P tile [q][key]

    const int tid = threadIdx.x, lane = tid & 31, wv = tid >> 5;
    const int col = lane & 15, khalf = lane >> 4;

    const int idx = blockIdx.x;
    const int qt4 = idx & 15;
    const int h   = (idx >> 4) & 31;
    const int b   = idx >> 9;
    const int kvh = h >> 2;
    const int qt  = qt4 * 4 + wv;

    const int nkb_w   = qt / 2 + 1;
    const int nkb_blk = 2 * qt4 + 2;

    const float slope = exp2f(-0.25f * (float)(h + 1));  // NH=32 pow2 ALiBi
    const float scale = 0.08838834764831845f;            // D^-0.5

    Frag qf[4];
    {
        const unsigned short* qb =
            q_ws + ((size_t)((b * NH + h) * SEQ + qt * 16 + col)) * DHEAD;
#pragma unroll
        for (int ks = 0; ks < 4; ++ks)
#pragma unroll
            for (int v = 0; v < 8; ++v)
                qf[ks].u[v] = *(const unsigned int*)&qb[ks * 32 + KA(v) + 8 * khalf];
    }

    floatx8 oa[8];
    float m_run[8], l_run[8];
#pragma unroll
    for (int nt = 0; nt < 8; ++nt)
#pragma unroll
        for (int v = 0; v < 8; ++v) oa[nt][v] = 0.0f;
#pragma unroll
    for (int v = 0; v < 8; ++v) { m_run[v] = -INFINITY; l_run[v] = 0.0f; }

    for (int kb = 0; kb < nkb_blk; ++kb) {
        const unsigned short* kbs =
            k_ws + ((size_t)((b * NKV + kvh) * SEQ + kb * 32)) * DHEAD;
        const unsigned short* vbs =
            v_ws + ((size_t)((b * NKV + kvh) * SEQ + kb * 32)) * DHEAD;
#pragma unroll
        for (int it = 0; it < 4; ++it) {
            int g = it * 128 + tid;
            int kl = g >> 4, c8 = (g & 15) << 3;
            async_copy_b128(&Kl[kl][c8], &kbs[kl * DHEAD + c8]);  // direct copy
            uint4 vv4 = *(const uint4*)&vbs[kl * DHEAD + c8];     // transpose V
            const unsigned short* vs = (const unsigned short*)&vv4;
#pragma unroll
            for (int j = 0; j < 8; ++j) Vl[c8 + j][kl] = vs[j];
        }
        wait_async0();
        __syncthreads();

        if (kb < nkb_w) {
            // scores S = Q K^T : two 16x16 tiles over 32 keys
            floatx8 s[2];
#pragma unroll
            for (int t = 0; t < 2; ++t)
#pragma unroll
                for (int v = 0; v < 8; ++v) s[t][v] = 0.0f;
#pragma unroll
            for (int ks = 0; ks < 4; ++ks) {
#pragma unroll
                for (int t = 0; t < 2; ++t) {
                    Frag bk;
#pragma unroll
                    for (int v = 0; v < 8; ++v)
                        bk.u[v] = *(const unsigned int*)
                            &Kl[t * 16 + col][ks * 32 + KB(v) + 8 * khalf];
                    s[t] = wmma_bf16(qf[ks], bk, s[t]);
                }
            }

            // ALiBi-sqrt bias + causal mask + block row max
            float mblk[8];
#pragma unroll
            for (int v = 0; v < 8; ++v) {
                const int qrow = qt * 16 + v + 8 * khalf;
                float mv = -INFINITY;
#pragma unroll
                for (int t = 0; t < 2; ++t) {
                    const int key = kb * 32 + t * 16 + col;
                    const int rel = qrow - key;
                    float sv = (rel >= 0)
                                   ? s[t][v] * scale - slope * sqrtf((float)rel)
                                   : -INFINITY;
                    s[t][v] = sv;
                    mv = fmaxf(mv, sv);
                }
                for (int off = 1; off <= 8; off <<= 1)
                    mv = fmaxf(mv, __shfl_xor(mv, off, 32));
                mblk[v] = mv;
            }

            // online softmax rescale
            float fac[8];
#pragma unroll
            for (int v = 0; v < 8; ++v) {
                float mnew = fmaxf(m_run[v], mblk[v]);
                fac[v] = __expf(m_run[v] - mnew);
                m_run[v] = mnew;
                l_run[v] *= fac[v];
            }
#pragma unroll
            for (int nt = 0; nt < 8; ++nt)
#pragma unroll
                for (int v = 0; v < 8; ++v) oa[nt][v] *= fac[v];

            // P = exp(S - m), row sums
#pragma unroll
            for (int v = 0; v < 8; ++v) {
                float ps = 0.0f;
#pragma unroll
                for (int t = 0; t < 2; ++t) {
                    float p = __expf(s[t][v] - m_run[v]);
                    s[t][v] = p;
                    ps += p;
                }
                for (int off = 1; off <= 8; off <<= 1)
                    ps += __shfl_xor(ps, off, 32);
                l_run[v] += ps;
            }

            // re-layout P (D-layout -> A-fragment) via wave-private LDS tile
#pragma unroll
            for (int t = 0; t < 2; ++t)
#pragma unroll
                for (int v = 0; v < 8; ++v)
                    Pl[wv][v + 8 * khalf][t * 16 + col] = f2bf(s[t][v]);

            Frag pa;
#pragma unroll
            for (int v = 0; v < 8; ++v)
                pa.u[v] = *(const unsigned int*)&Pl[wv][col][KA(v) + 8 * khalf];

            // O += P V : 8 n-tiles over D=128, K-dim = 32 keys
#pragma unroll
            for (int nt = 0; nt < 8; ++nt) {
                Frag bv;
#pragma unroll
                for (int v = 0; v < 8; ++v)
                    bv.u[v] = *(const unsigned int*)
                        &Vl[nt * 16 + col][KB(v) + 8 * khalf];
                oa[nt] = wmma_bf16(pa, bv, oa[nt]);
            }
        }
        __syncthreads();
    }

    // normalize and write bf16 attention output [b][s][h*128+d]
#pragma unroll
    for (int v = 0; v < 8; ++v) {
        float inv = 1.0f / l_run[v];
        int srow = qt * 16 + v + 8 * khalf;
        unsigned short* orow =
            o_ws + ((size_t)(b * SEQ + srow)) * (NH * DHEAD) + h * DHEAD;
#pragma unroll
        for (int nt = 0; nt < 8; ++nt)
            orow[nt * 16 + col] = f2bf(oa[nt][v] * inv);
    }
}

// ---------------------------------------------------------------------------
// Host launcher
// ---------------------------------------------------------------------------
extern "C" void kernel_launch(void* const* d_in, const int* in_sizes, int n_in,
                              void* d_out, int out_size, void* d_ws, size_t ws_size,
                              hipStream_t stream) {
    (void)in_sizes; (void)n_in; (void)out_size; (void)ws_size;

    const float* hidden = (const float*)d_in[0];
    const float* w_qkv  = (const float*)d_in[1];
    const float* w_o    = (const float*)d_in[2];
    float* out = (float*)d_out;

    char* ws = (char*)d_ws;
    const size_t Q_BYTES   = (size_t)BATCH * NH  * SEQ * DHEAD * 2;  // 32 MiB
    const size_t K_BYTES   = (size_t)BATCH * NKV * SEQ * DHEAD * 2;  //  8 MiB
    const size_t V_BYTES   = K_BYTES;                                //  8 MiB
    const size_t WQT_BYTES = (size_t)QKV_N * HDIM * 2;               // 48 MiB
    const size_t WOT_BYTES = (size_t)HDIM * HDIM * 2;                // 32 MiB

    unsigned short* q_ws = (unsigned short*)(ws);
    unsigned short* k_ws = (unsigned short*)(ws + Q_BYTES);
    unsigned short* v_ws = (unsigned short*)(ws + Q_BYTES + K_BYTES);
    unsigned short* wqT  = (unsigned short*)(ws + Q_BYTES + K_BYTES + V_BYTES);
    unsigned short* woT  = (unsigned short*)(ws + Q_BYTES + K_BYTES + V_BYTES + WQT_BYTES);
    // hidden-bf16 aliases attn-out: gemm_qkv fully consumes it before attn writes
    unsigned short* h_bf = (unsigned short*)(ws + Q_BYTES + K_BYTES + V_BYTES +
                                             WQT_BYTES + WOT_BYTES);
    unsigned short* o_ws = h_bf;

    // 0) one-time precision prep (bf16 everywhere in the hot loops)
    cvt_bf16_kernel<<<dim3((MROWS * HDIM / 4 + 255) / 256), 256, 0, stream>>>(
        hidden, h_bf, MROWS * HDIM / 4);
    cvt_transpose_kernel<<<dim3(QKV_N / 32, HDIM / 32), 256, 0, stream>>>(
        w_qkv, wqT, HDIM, QKV_N);
    cvt_transpose_kernel<<<dim3(HDIM / 32, HDIM / 32), 256, 0, stream>>>(
        w_o, woT, HDIM, HDIM);

    // 1) QKV projection: grid (6144/128, 4096/128)
    gemm_qkv_kernel<<<dim3(QKV_N / 128, MROWS / 128), 256, 0, stream>>>(
        h_bf, wqT, q_ws, k_ws, v_ws);

    // 2) flash attention: B * NH * (S/64) blocks, 4 waves each
    attn_kernel<<<dim3(BATCH * NH * (SEQ / 64)), 128, 0, stream>>>(
        q_ws, k_ws, v_ws, o_ws);

    // 3) output projection: grid (4096/128, 4096/128)
    gemm_o_kernel<<<dim3(HDIM / 128, MROWS / 128), 256, 0, stream>>>(
        o_ws, woT, out);
}
